// Attention_18459769439020
// MI455X (gfx1250) — matrix operand order repaired and checked by reading, not compiled, not run on hardware
//
#include <hip/hip_runtime.h>
#include <math.h>

// Problem constants (match reference)
#define B_ 2
#define S_ 2048
#define E_ 1024
#define H_ 16
#define D_ 64

typedef __attribute__((ext_vector_type(16))) __bf16 v16bf;
typedef __attribute__((ext_vector_type(8)))  __bf16 v8bf;
typedef __attribute__((ext_vector_type(8)))  float  v8f;

// ---------------------------------------------------------------------------
// WMMA fragment helpers (CDNA5 16x16x32 bf16, wave32)
//
// A operand (16x32, MxK): lane L holds row M = L&15.
//   half = L>>4 :  vector elems 0..7  <-> K = 8*half + 0..7
//                  vector elems 8..15 <-> K = 16 + 8*half + 0..7
// B operand (32x16, KxN): lane L holds column N = L&15.
//   half = L>>4 :  same K mapping as A (elems 0..7 <-> 8*half, 8..15 <-> 16 + 8*half)
//   => column n of B == row n of B^T, so we feed B from [N,K]-major storage.
// C/D (16x16 f32): VGPR r, lane L: M = r + 8*(L>>4), N = L&15.
// ---------------------------------------------------------------------------

static __device__ __forceinline__ v16bf load_frag_A(const __bf16* rowK0, int half) {
  v16bf a;
  *(v8bf*)&a       = *(const v8bf*)(rowK0 + 8 * half);
  *((v8bf*)&a + 1) = *(const v8bf*)(rowK0 + 16 + 8 * half);
  return a;
}

static __device__ __forceinline__ v16bf load_frag_B(const __bf16* rowNK0, int half) {
  v16bf b;
  *(v8bf*)&b       = *(const v8bf*)(rowNK0 + 8 * half);
  *((v8bf*)&b + 1) = *(const v8bf*)(rowNK0 + 16 + 8 * half);
  return b;
}

static __device__ __forceinline__ v8f wmma_bf16(v16bf a, v16bf b, v8f c) {
  return __builtin_amdgcn_wmma_f32_16x16x32_bf16(
      /*neg_a=*/false, a, /*neg_b=*/false, b,
      /*c_mod=*/(short)0, c, /*reuse_a=*/false, /*reuse_b=*/false);
}

// ---------------------------------------------------------------------------
// Elementwise converts / transposes
// ---------------------------------------------------------------------------

__global__ void cvt_f32_to_bf16(const float* __restrict__ in,
                                __bf16* __restrict__ out, int n) {
  int i = blockIdx.x * blockDim.x + threadIdx.x;
  if (i < n) out[i] = (__bf16)in[i];
}

// Wt[n*E + k] = (bf16) W[k*E + n]   (E x E)
__global__ void transpose_w(const float* __restrict__ W,
                            __bf16* __restrict__ Wt) {
  int o = blockIdx.x * blockDim.x + threadIdx.x;
  if (o >= E_ * E_) return;
  int k = o & (E_ - 1);
  int n = o >> 10;           // E_ = 1024 = 2^10
  Wt[o] = (__bf16)W[(size_t)k * E_ + n];
}

// Vt[((b*H + h)*D + d)*S + s] = V[(b*S + s)*E + h*D + d]
__global__ void transpose_v(const __bf16* __restrict__ Vb,
                            __bf16* __restrict__ Vt) {
  size_t o = (size_t)blockIdx.x * blockDim.x + threadIdx.x;
  if (o >= (size_t)B_ * H_ * D_ * S_) return;
  int s = (int)(o & (S_ - 1));          // 2^11
  int d = (int)((o >> 11) & (D_ - 1));  // 2^6
  int h = (int)((o >> 17) & (H_ - 1));  // 2^4
  int b = (int)(o >> 21);
  Vt[o] = Vb[((size_t)b * S_ + s) * E_ + h * D_ + d];
}

// ---------------------------------------------------------------------------
// Generic GEMM:  C[M,N] = A[M,K] @ B[K,N] + bias,  B given as Bt[N,K].
// One wave computes a 16(M) x 64(N) tile; 8 waves / block.
// Software-pipelined: next K-step fragments are loaded before the current
// K-step's WMMAs issue, so waits become partial (overlap loads with math).
// ---------------------------------------------------------------------------

template <bool OUT_F32>
__global__ __launch_bounds__(256) void gemm_bf16(
    const __bf16* __restrict__ A,   // [M,K]
    const __bf16* __restrict__ Bt,  // [N,K]
    const float* __restrict__ bias, // [N] or nullptr
    void* __restrict__ Cout,        // f32 or bf16 [M,N]
    int M, int N, int K) {
  const int lane  = threadIdx.x & 31;
  const int wave  = threadIdx.x >> 5;
  const int gwave = blockIdx.x * (blockDim.x >> 5) + wave;
  const int tilesN = N >> 6;                 // 64-wide wave tiles
  const int m0 = (gwave / tilesN) * 16;
  const int n0 = (gwave % tilesN) * 64;
  if (m0 >= M) return;

  const int half = lane >> 4;
  const int l15  = lane & 15;

  const __bf16* arow = A + (size_t)(m0 + l15) * K;
  const __bf16* brow[4];
#pragma unroll
  for (int t = 0; t < 4; ++t)
    brow[t] = Bt + (size_t)(n0 + t * 16 + l15) * K;

  v8f acc[4] = {};

  // prologue
  v16bf a_cur = load_frag_A(arow, half);
  v16bf b_cur[4];
#pragma unroll
  for (int t = 0; t < 4; ++t) b_cur[t] = load_frag_B(brow[t], half);

  for (int k0 = 32; k0 < K; k0 += 32) {
    // issue next-step loads first (no deps on current WMMA inputs)
    v16bf a_nxt = load_frag_A(arow + k0, half);
    v16bf b_nxt[4];
#pragma unroll
    for (int t = 0; t < 4; ++t) b_nxt[t] = load_frag_B(brow[t] + k0, half);

    // prefetch two K-steps ahead (global_prefetch_b8)
    if (k0 + 64 < K) {
      __builtin_prefetch(arow + k0 + 64, 0, 1);
#pragma unroll
      for (int t = 0; t < 4; ++t) __builtin_prefetch(brow[t] + k0 + 64, 0, 1);
    }

#pragma unroll
    for (int t = 0; t < 4; ++t) acc[t] = wmma_bf16(a_cur, b_cur[t], acc[t]);

    a_cur = a_nxt;
#pragma unroll
    for (int t = 0; t < 4; ++t) b_cur[t] = b_nxt[t];
  }
  // epilogue
#pragma unroll
  for (int t = 0; t < 4; ++t) acc[t] = wmma_bf16(a_cur, b_cur[t], acc[t]);

#pragma unroll
  for (int t = 0; t < 4; ++t) {
    const int n = n0 + t * 16 + l15;
    const float bv = bias ? bias[n] : 0.0f;
#pragma unroll
    for (int r = 0; r < 8; ++r) {
      const int m = m0 + r + 8 * half;
      float v = acc[t][r] + bv;
      if (OUT_F32)
        ((float*)Cout)[(size_t)m * N + n] = v;
      else
        ((__bf16*)Cout)[(size_t)m * N + n] = (__bf16)v;
    }
  }
}

// ---------------------------------------------------------------------------
// scores + softmax: one block = one (b,h) and 16 query rows.
// 8 waves; wave w covers keys [w*256, w*256+256). Scores live in dynamic LDS
// (16 x (S+8) f32 ~ 128KB, fine under 320KB/WGP), softmax reduced in LDS,
// probs normalized in place, then streamed to global via the async DMA path
// (GLOBAL_STORE_ASYNC_FROM_LDS_B128, tracked on ASYNCcnt).
// ---------------------------------------------------------------------------

#define LDS_PAD 8
#define LDSW (S_ + LDS_PAD)

__global__ __launch_bounds__(256) void scores_softmax(
    const __bf16* __restrict__ Qb,  // [B*S, E]
    const __bf16* __restrict__ Kb,  // [B*S, E]
    float* __restrict__ probs) {    // [B,H,S,S]
  extern __shared__ float lds[];
  float* sc  = lds;                 // [16][LDSW]
  float* red = lds + 16 * LDSW;     // [16][17]

  const int bh = blockIdx.x / (S_ / 16);
  const int qt = blockIdx.x % (S_ / 16);
  const int b  = bh / H_;
  const int h  = bh % H_;

  const int lane = threadIdx.x & 31;
  const int wave = threadIdx.x >> 5;
  const int half = lane >> 4;
  const int l15  = lane & 15;

  // A fragments: Q rows qt*16 + (0..15), head slice h*D .. h*D+63
  const __bf16* qrow = Qb + ((size_t)b * S_ + qt * 16 + l15) * E_ + h * D_;
  v16bf aq0 = load_frag_A(qrow + 0, half);
  v16bf aq1 = load_frag_A(qrow + 32, half);

  // key row pointer for this lane's B-column; wave covers 256 keys
  const __bf16* kbase = Kb + ((size_t)b * S_ + wave * 256 + l15) * E_ + h * D_;
  const size_t krow_step = (size_t)16 * E_;   // 16 key rows per tile

  const float scale = 0.125f;  // 1/sqrt(64)

  // prologue: tile 0's B fragments
  v16bf bk0 = load_frag_B(kbase + 0, half);
  v16bf bk1 = load_frag_B(kbase + 32, half);

  for (int t = 0; t < 16; ++t) {
    // double-buffer: issue tile t+1's loads before tile t's WMMAs
    v16bf nb0, nb1;
    if (t < 15) {
      const __bf16* krow = kbase + (size_t)(t + 1) * krow_step;
      nb0 = load_frag_B(krow + 0, half);
      nb1 = load_frag_B(krow + 32, half);
      if (t < 14) __builtin_prefetch(kbase + (size_t)(t + 2) * krow_step, 0, 1);
    }

    v8f acc = {};
    acc = wmma_bf16(aq0, bk0, acc);
    acc = wmma_bf16(aq1, bk1, acc);

    const int kkey = wave * 256 + t * 16;
#pragma unroll
    for (int r = 0; r < 8; ++r) {
      const int m = r + 8 * half;
      sc[m * LDSW + kkey + l15] = acc[r] * scale;
    }
    if (t < 15) { bk0 = nb0; bk1 = nb1; }
  }
  __syncthreads();

  // ---- softmax over each of 16 rows (16 threads x 128 elems per row) ----
  const int r   = threadIdx.x >> 4;
  const int seg = threadIdx.x & 15;
  float* row = sc + r * LDSW + seg * 128;

  float mx = -3.0e38f;
  for (int i = 0; i < 128; ++i) mx = fmaxf(mx, row[i]);
  red[r * 17 + seg] = mx;
  __syncthreads();
  float rmx = red[r * 17];
  for (int i = 1; i < 16; ++i) rmx = fmaxf(rmx, red[r * 17 + i]);
  __syncthreads();

  float s = 0.0f;
  for (int i = 0; i < 128; ++i) {
    float e = __expf(row[i] - rmx);
    row[i] = e;
    s += e;
  }
  red[r * 17 + seg] = s;
  __syncthreads();
  if (seg == 0) {
    float tot = 0.0f;
    for (int i = 0; i < 16; ++i) tot += red[r * 17 + i];
    red[r * 17 + 16] = 1.0f / tot;
  }
  __syncthreads();

  // ---- normalize in place in LDS ----
  const float inv = red[r * 17 + 16];
  for (int i = 0; i < 128; ++i) row[i] *= inv;
  __syncthreads();

  // ---- async DMA writeout: LDS -> global, 16B per lane per issue ----
  // 16*2048 floats = 32 iterations x (256 lanes x 4 floats).
  float* outp = probs + ((size_t)bh * S_ + (size_t)qt * 16) * S_;
  for (int it = 0; it < 32; ++it) {
    const int e  = it * 1024 + (int)threadIdx.x * 4;  // flat f32 index
    const int rr = e >> 11;                           // / S_
    const int cc = e & (S_ - 1);
    // byte offset of source within this workgroup's (all-dynamic) LDS
    const unsigned loff = (unsigned)(((size_t)rr * LDSW + cc) * sizeof(float));
    const float* gp = outp + (size_t)rr * S_ + cc;
    asm volatile("global_store_async_from_lds_b128 %0, %1, off"
                 :: "v"(gp), "v"(loff) : "memory");
  }
  asm volatile("s_wait_asynccnt 0x0" ::: "memory");
}

// ---------------------------------------------------------------------------
// ctx = probs @ V :  per (b,h)  M=S (16-row wave tiles), N=D=64, K=S.
// probs (f32) converted to bf16 in-registers for the A operand; V supplied
// transposed (Vt[b,h,d,s]) so B operand rows are contiguous. Pipelined.
// Output stored interleaved as ctx[b, s, h*D + d] (bf16) for the out-proj.
// ---------------------------------------------------------------------------

static __device__ __forceinline__ v16bf cvt_prob_frag(v8f p0, v8f p1) {
  v16bf a;
#pragma unroll
  for (int i = 0; i < 8; ++i) {
    a[i]     = (__bf16)p0[i];
    a[i + 8] = (__bf16)p1[i];
  }
  return a;
}

__global__ __launch_bounds__(256) void ctx_gemm(
    const float* __restrict__ probs,  // [B,H,S,S]
    const __bf16* __restrict__ Vt,    // [B,H,D,S]
    __bf16* __restrict__ ctx) {       // [B,S,E]
  const int lane = threadIdx.x & 31;
  const int wave = threadIdx.x >> 5;
  const int bh   = blockIdx.x / (S_ / 128);
  const int qblk = blockIdx.x % (S_ / 128);
  const int b    = bh / H_;
  const int h    = bh % H_;
  const int m0   = (qblk * 8 + wave) * 16;

  const int half = lane >> 4;
  const int l15  = lane & 15;

  const float*  prow  = probs + ((size_t)bh * S_ + m0 + l15) * S_;
  const __bf16* vbase = Vt + (size_t)bh * D_ * S_;
  const __bf16* vrow[4];
#pragma unroll
  for (int t = 0; t < 4; ++t)
    vrow[t] = vbase + (size_t)(t * 16 + l15) * S_;

  v8f acc[4] = {};

  // prologue
  v8f p0c = *(const v8f*)(prow + 8 * half);
  v8f p1c = *(const v8f*)(prow + 16 + 8 * half);
  v16bf b_cur[4];
#pragma unroll
  for (int t = 0; t < 4; ++t) b_cur[t] = load_frag_B(vrow[t], half);

  for (int k0 = 32; k0 < S_; k0 += 32) {
    // next-step loads first
    v8f p0n = *(const v8f*)(prow + k0 + 8 * half);
    v8f p1n = *(const v8f*)(prow + k0 + 16 + 8 * half);
    v16bf b_nxt[4];
#pragma unroll
    for (int t = 0; t < 4; ++t) b_nxt[t] = load_frag_B(vrow[t] + k0, half);

    if (k0 + 64 < S_) {
      __builtin_prefetch(prow + k0 + 64, 0, 1);
#pragma unroll
      for (int t = 0; t < 4; ++t) __builtin_prefetch(vrow[t] + k0 + 64, 0, 1);
    }

    v16bf a = cvt_prob_frag(p0c, p1c);
#pragma unroll
    for (int t = 0; t < 4; ++t) acc[t] = wmma_bf16(a, b_cur[t], acc[t]);

    p0c = p0n; p1c = p1n;
#pragma unroll
    for (int t = 0; t < 4; ++t) b_cur[t] = b_nxt[t];
  }
  // epilogue
  {
    v16bf a = cvt_prob_frag(p0c, p1c);
#pragma unroll
    for (int t = 0; t < 4; ++t) acc[t] = wmma_bf16(a, b_cur[t], acc[t]);
  }

#pragma unroll
  for (int t = 0; t < 4; ++t) {
    const int d = t * 16 + l15;
#pragma unroll
    for (int r = 0; r < 8; ++r) {
      const int m = m0 + r + 8 * half;
      ctx[((size_t)b * S_ + m) * E_ + h * D_ + d] = (__bf16)acc[t][r];
    }
  }
}

// ---------------------------------------------------------------------------
// Host-side launch
// ---------------------------------------------------------------------------

extern "C" void kernel_launch(void* const* d_in, const int* in_sizes, int n_in,
                              void* d_out, int out_size, void* d_ws, size_t ws_size,
                              hipStream_t stream) {
  const float* hidden = (const float*)d_in[0];
  const float* Wq = (const float*)d_in[1];
  const float* bq = (const float*)d_in[2];
  const float* Wk = (const float*)d_in[3];
  const float* bk = (const float*)d_in[4];
  const float* Wv = (const float*)d_in[5];
  const float* bv = (const float*)d_in[6];
  const float* Wo = (const float*)d_in[7];
  const float* bo = (const float*)d_in[8];

  const size_t NTOK = (size_t)B_ * S_;          // 4096
  const size_t XE   = NTOK * E_;                // 4,194,304
  const size_t WE   = (size_t)E_ * E_;          // 1,048,576

  // Workspace carve-out (bf16 buffers), 256B aligned
  size_t off = 0;
  auto carve = [&](size_t bytes) -> void* {
    void* p = (char*)d_ws + off;
    off += (bytes + 255) & ~(size_t)255;
    return p;
  };
  __bf16* Xb  = (__bf16*)carve(XE * 2);
  __bf16* Wqt = (__bf16*)carve(WE * 2);
  __bf16* Wkt = (__bf16*)carve(WE * 2);
  __bf16* Wvt = (__bf16*)carve(WE * 2);
  __bf16* Wot = (__bf16*)carve(WE * 2);
  __bf16* Qb  = (__bf16*)carve(XE * 2);
  __bf16* Kb  = (__bf16*)carve(XE * 2);
  __bf16* Vb  = (__bf16*)carve(XE * 2);
  __bf16* Vt  = (__bf16*)carve(XE * 2);
  __bf16* Ctx = (__bf16*)carve(XE * 2);
  (void)ws_size;

  float* out_f32 = (float*)d_out;                       // [B,S,E]
  float* probs   = (float*)d_out + XE;                  // [B,H,S,S]

  // 1) converts / weight transposes
  cvt_f32_to_bf16<<<(int)((XE + 255) / 256), 256, 0, stream>>>(hidden, Xb, (int)XE);
  transpose_w<<<(int)((WE + 255) / 256), 256, 0, stream>>>(Wq, Wqt);
  transpose_w<<<(int)((WE + 255) / 256), 256, 0, stream>>>(Wk, Wkt);
  transpose_w<<<(int)((WE + 255) / 256), 256, 0, stream>>>(Wv, Wvt);
  transpose_w<<<(int)((WE + 255) / 256), 256, 0, stream>>>(Wo, Wot);

  // 2) Q/K/V projections: M=4096, N=1024, K=1024
  const int gemm_waves  = (int)((NTOK / 16) * (E_ / 64));  // 4096
  const int gemm_blocks = gemm_waves / 8;                  // 512
  gemm_bf16<false><<<gemm_blocks, 256, 0, stream>>>(Xb, Wqt, bq, Qb, (int)NTOK, E_, E_);
  gemm_bf16<false><<<gemm_blocks, 256, 0, stream>>>(Xb, Wkt, bk, Kb, (int)NTOK, E_, E_);
  gemm_bf16<false><<<gemm_blocks, 256, 0, stream>>>(Xb, Wvt, bv, Vb, (int)NTOK, E_, E_);

  // 3) transpose V -> [B,H,D,S]
  transpose_v<<<(int)((XE + 255) / 256), 256, 0, stream>>>(Vb, Vt);

  // 4) scores + softmax -> probs (async LDS->global writeout)
  const size_t smem = (size_t)(16 * LDSW + 16 * 17) * sizeof(float);  // ~132KB
  hipFuncSetAttribute((const void*)scores_softmax,
                      hipFuncAttributeMaxDynamicSharedMemorySize, (int)smem);
  scores_softmax<<<B_ * H_ * (S_ / 16), 256, smem, stream>>>(Qb, Kb, probs);

  // 5) ctx = probs @ V
  ctx_gemm<<<B_ * H_ * (S_ / 128), 256, 0, stream>>>(probs, Vt, Ctx);

  // 6) out = ctx @ Wo + bo  (f32 to d_out head)
  gemm_bf16<true><<<gemm_blocks, 256, 0, stream>>>(Ctx, Wot, bo, out_f32, (int)NTOK, E_, E_);
}
